// ToyMoE_65721589563787
// MI455X (gfx1250) — matrix operand order repaired and checked by reading, not compiled, not run on hardware
//
#include <hip/hip_runtime.h>

#define NTOK 16384
#define DDIM 1024
#define NEXP 8
#define APITCH 1032   // 1024 bf16 + 8 pad -> row step = 2064B -> bank stride 4, conflict-free

typedef __attribute__((ext_vector_type(16))) __bf16 v16bf;
typedef __attribute__((ext_vector_type(8)))  float  v8f;

union FragB { v16bf v; uint4 u[2]; };

__device__ __forceinline__ unsigned short f2bf(float x) {
    unsigned int u = __float_as_uint(x);
    unsigned int r = 0x7FFFu + ((u >> 16) & 1u);   // round to nearest even
    return (unsigned short)((u + r) >> 16);
}
__device__ __forceinline__ unsigned int pack2(float a, float b) {
    return (unsigned int)f2bf(a) | ((unsigned int)f2bf(b) << 16);
}

// ---------------- 1. zero per-expert counters ----------------
__global__ void zcnt_kernel(int* counts) {
    if (threadIdx.x < NEXP) counts[threadIdx.x] = 0;
}

// ---------------- 2. router: logits, softmax, top-2; fused x -> bf16 ----------------
__global__ __launch_bounds__(256) void router_kernel(
    const float* __restrict__ x, const float* __restrict__ Wr,
    uint2* __restrict__ xbf4,            // 4 bf16 per uint2, row pitch D/4
    int2* __restrict__ topi, float2* __restrict__ topv)
{
    const int wave = threadIdx.x >> 5;
    const int lane = threadIdx.x & 31;
    const int tok  = blockIdx.x * 8 + wave;

    const float4* xr = (const float4*)(x + (size_t)tok * DDIM);
    float acc[NEXP];
#pragma unroll
    for (int e = 0; e < NEXP; ++e) acc[e] = 0.f;

#pragma unroll
    for (int i = 0; i < 8; ++i) {
        const int d4 = lane + i * 32;              // float4 index within row
        float4 xv = xr[d4];
        uint2 p; p.x = pack2(xv.x, xv.y); p.y = pack2(xv.z, xv.w);
        xbf4[(size_t)tok * (DDIM / 4) + d4] = p;   // bf16 copy of x
#pragma unroll
        for (int e = 0; e < NEXP; ++e) {
            float4 wv = ((const float4*)(Wr + (size_t)e * DDIM))[d4];
            acc[e] += xv.x * wv.x + xv.y * wv.y + xv.z * wv.z + xv.w * wv.w;
        }
    }
    // wave32 butterfly reduction: every lane ends with the full sums
#pragma unroll
    for (int off = 16; off >= 1; off >>= 1)
#pragma unroll
        for (int e = 0; e < NEXP; ++e)
            acc[e] += __shfl_xor(acc[e], off, 32);

    if (lane == 0) {
        float mx = acc[0];
#pragma unroll
        for (int e = 1; e < NEXP; ++e) mx = fmaxf(mx, acc[e]);
        float g[NEXP]; float s = 0.f;
#pragma unroll
        for (int e = 0; e < NEXP; ++e) { g[e] = __expf(acc[e] - mx); s += g[e]; }
        const float inv = 1.f / s;
#pragma unroll
        for (int e = 0; e < NEXP; ++e) g[e] *= inv;
        int i0 = 0; float v0 = g[0];
#pragma unroll
        for (int e = 1; e < NEXP; ++e) if (g[e] > v0) { v0 = g[e]; i0 = e; }
        int i1 = (i0 == 0) ? 1 : 0; float v1 = g[i1];
#pragma unroll
        for (int e = 0; e < NEXP; ++e) if (e != i0 && g[e] > v1) { v1 = g[e]; i1 = e; }
        topi[tok] = make_int2(i0, i1);
        topv[tok] = make_float2(v0, v1);
    }
}

// ---------------- 3. We f32 [e][k][n] -> bf16 transposed [e][n][k] -----------------
// Strided reads hit L2 (We fits in 192MB L2, each line re-referenced 16x);
// writes are fully coalesced b128.
__global__ __launch_bounds__(256) void wconv_kernel(
    const float* __restrict__ We, unsigned short* __restrict__ wbfT)
{
    const int gid = blockIdx.x * 256 + threadIdx.x;        // E*D*D/8 threads
    const int e   = gid / (DDIM * (DDIM / 8));
    const int rem = gid % (DDIM * (DDIM / 8));
    const int n   = rem / (DDIM / 8);
    const int kc  = (rem % (DDIM / 8)) * 8;
    const float* src = We + ((size_t)e * DDIM + kc) * DDIM + n;
    float v0 = src[0];
    float v1 = src[DDIM];
    float v2 = src[2 * DDIM];
    float v3 = src[3 * DDIM];
    float v4 = src[4 * DDIM];
    float v5 = src[5 * DDIM];
    float v6 = src[6 * DDIM];
    float v7 = src[7 * DDIM];
    uint4 p;
    p.x = pack2(v0, v1); p.y = pack2(v2, v3);
    p.z = pack2(v4, v5); p.w = pack2(v6, v7);
    *(uint4*)(wbfT + ((size_t)e * DDIM + n) * DDIM + kc) = p;
}

// ---------------- 4. build per-expert token lists ----------------
__global__ __launch_bounds__(256) void scatter_kernel(
    const int2* __restrict__ topi, const float2* __restrict__ topv,
    int* __restrict__ counts, int* __restrict__ lists, float* __restrict__ lw)
{
    const int n = blockIdx.x * 256 + threadIdx.x;
    if (n >= NTOK) return;
    int2 ti = topi[n]; float2 tv = topv[n];
    int p0 = atomicAdd(&counts[ti.x], 1);
    lists[ti.x * NTOK + p0] = n; lw[ti.x * NTOK + p0] = tv.x;
    int p1 = atomicAdd(&counts[ti.y], 1);
    lists[ti.y * NTOK + p1] = n; lw[ti.y * NTOK + p1] = tv.y;
}

// ---------------- 5. out = topv0*be[e0] + topv1*be[e1]  (init, un-poisons d_out) ---
__global__ __launch_bounds__(256) void bias_kernel(
    const int2* __restrict__ topi, const float2* __restrict__ topv,
    const float* __restrict__ be, float* __restrict__ out)
{
    const int tok = blockIdx.x;
    const int d   = threadIdx.x * 4;
    int2 ti = topi[tok]; float2 tv = topv[tok];
    float4 b0 = *(const float4*)(be + (size_t)ti.x * DDIM + d);
    float4 b1 = *(const float4*)(be + (size_t)ti.y * DDIM + d);
    float4 o;
    o.x = tv.x * b0.x + tv.y * b1.x;
    o.y = tv.x * b0.y + tv.y * b1.y;
    o.z = tv.x * b0.z + tv.y * b1.z;
    o.w = tv.x * b0.w + tv.y * b1.w;
    *(float4*)(out + (size_t)tok * DDIM + d) = o;
}

// ---------------- 6. gathered grouped GEMM: bf16 WMMA, f32 accumulate --------------
// block = 128 thr (4 waves); block tile = 32 tokens x 256 out-cols.
// Each wave: 32x64 (two 16-row A fragments share every B fragment -> 2x B reuse).
__global__ __launch_bounds__(128) void moe_gemm_kernel(
    const unsigned short* __restrict__ xbf,
    const unsigned short* __restrict__ wbfT,
    const int*   __restrict__ counts,
    const int*   __restrict__ lists,
    const float* __restrict__ lw,
    float* __restrict__ out)
{
    const int e     = blockIdx.z;
    const int cnt   = counts[e];
    const int mbase = blockIdx.x * 32;
    if (mbase >= cnt) return;

    __shared__ __align__(16) unsigned short As[32 * APITCH];
    __shared__ int   toks[32];
    __shared__ float wts[32];

    const int tid = threadIdx.x;
    if (tid < 32) {
        int i   = mbase + tid;
        int idx = (i < cnt) ? i : cnt - 1;           // clamp: never read garbage ids
        toks[tid] = lists[e * NTOK + idx];
        wts[tid]  = (i < cnt) ? lw[e * NTOK + idx] : 0.f;
    }
    __syncthreads();

    // stage 32 gathered bf16 rows (32 x 1024) into LDS, b128 chunks
    for (int j = 0; j < 32; ++j) {
        const uint4* src = (const uint4*)(xbf + (size_t)toks[j] * DDIM) + tid;
        *(uint4*)&As[j * APITCH + tid * 8] = *src;
    }
    __syncthreads();

    const int lane  = tid & 31;
    const int wid   = tid >> 5;
    const int sel   = lane >> 4;                     // half-wave select
    const int l16   = lane & 15;
    const int nbase = blockIdx.y * 256 + wid * 64;

    v8f acc[2][4];
#pragma unroll
    for (int h = 0; h < 2; ++h)
#pragma unroll
        for (int j = 0; j < 4; ++j) acc[h][j] = 0.0f;

    const unsigned short* wb = wbfT + (size_t)e * DDIM * DDIM;

    for (int k0 = 0; k0 < DDIM; k0 += 32) {
        // A fragments per ISA 16-bit 16x32 layout:
        // lanes<16: K [k0,k0+8) + [k0+16,k0+24); lanes>=16: +8 each
        FragB af0, af1;
        const unsigned short* ap0 = &As[l16 * APITCH + k0 + sel * 8];
        af0.u[0] = *(const uint4*)ap0;
        af0.u[1] = *(const uint4*)(ap0 + 16);
        const unsigned short* ap1 = ap0 + 16 * APITCH;
        af1.u[0] = *(const uint4*)ap1;
        af1.u[1] = *(const uint4*)(ap1 + 16);
#pragma unroll
        for (int j = 0; j < 4; ++j) {
            const int ncol = nbase + j * 16 + l16;
            // B fragment: lane holds column ncol, K [k0+8*sel,+8) + [k0+16+8*sel,+8) (transposed We)
            const uint4* bp = (const uint4*)(wb + (size_t)ncol * DDIM + k0 + sel * 8);
            FragB bf_; bf_.u[0] = bp[0]; bf_.u[1] = bp[2];
            acc[0][j] = __builtin_amdgcn_wmma_f32_16x16x32_bf16(
                false, af0.v, false, bf_.v, (short)0, acc[0][j], false, false);
            acc[1][j] = __builtin_amdgcn_wmma_f32_16x16x32_bf16(
                false, af1.v, false, bf_.v, (short)0, acc[1][j], false, false);
        }
    }

    // D layout: VGPR v -> row v + 8*sel, N = l16. Scale by gate weight, atomic add.
    // Uniform full-tile fast path: no per-element EXEC churn.
    if (mbase + 32 <= cnt) {
#pragma unroll
        for (int h = 0; h < 2; ++h) {
#pragma unroll
            for (int v = 0; v < 8; ++v) {
                const int m = h * 16 + sel * 8 + v;
                const float w = wts[m];
                float* p = out + (size_t)toks[m] * DDIM + nbase + l16;
#pragma unroll
                for (int j = 0; j < 4; ++j)
                    atomicAdd(p + j * 16, acc[h][j][v] * w);
            }
        }
    } else {
#pragma unroll
        for (int h = 0; h < 2; ++h) {
#pragma unroll
            for (int v = 0; v < 8; ++v) {
                const int m = h * 16 + sel * 8 + v;
                if (mbase + m < cnt) {
                    const float w = wts[m];
                    float* p = out + (size_t)toks[m] * DDIM + nbase + l16;
#pragma unroll
                    for (int j = 0; j < 4; ++j)
                        atomicAdd(p + j * 16, acc[h][j][v] * w);
                }
            }
        }
    }
}

// ---------------- launch ----------------
extern "C" void kernel_launch(void* const* d_in, const int* in_sizes, int n_in,
                              void* d_out, int out_size, void* d_ws, size_t ws_size,
                              hipStream_t stream) {
    const float* x  = (const float*)d_in[0];
    const float* Wr = (const float*)d_in[1];
    const float* We = (const float*)d_in[2];
    const float* be = (const float*)d_in[3];
    float* out = (float*)d_out;

    unsigned char* ws = (unsigned char*)d_ws;
    size_t off = 0;
    unsigned short* xbf  = (unsigned short*)(ws + off); off += (size_t)NTOK * DDIM * 2;          // 32 MB
    unsigned short* wbfT = (unsigned short*)(ws + off); off += (size_t)NEXP * DDIM * DDIM * 2;   // 16 MB
    int2*   topi   = (int2*)  (ws + off); off += (size_t)NTOK * 8;
    float2* topv   = (float2*)(ws + off); off += (size_t)NTOK * 8;
    int*    counts = (int*)   (ws + off); off += 256;
    int*    lists  = (int*)   (ws + off); off += (size_t)NEXP * NTOK * 4;
    float*  lw     = (float*) (ws + off); off += (size_t)NEXP * NTOK * 4;
    (void)off; (void)ws_size; (void)in_sizes; (void)n_in; (void)out_size;

    zcnt_kernel<<<1, 32, 0, stream>>>(counts);
    router_kernel<<<NTOK / 8, 256, 0, stream>>>(x, Wr, (uint2*)xbf, topi, topv);
    wconv_kernel<<<(NEXP * DDIM * (DDIM / 8)) / 256, 256, 0, stream>>>(We, wbfT);
    scatter_kernel<<<NTOK / 256, 256, 0, stream>>>(topi, topv, counts, lists, lw);
    bias_kernel<<<NTOK, 256, 0, stream>>>(topi, topv, be, out);
    dim3 gg(NTOK / 32, DDIM / 256, NEXP);   // over-provisioned in x; blocks early-exit on cnt
    moe_gemm_kernel<<<gg, 128, 0, stream>>>(xbf, wbfT, counts, lists, lw, out);
}
